// BackboneDenoisingLayer_8323646620402
// MI455X (gfx1250) — matrix operand order repaired and checked by reading, not compiled, not run on hardware
//
#include <hip/hip_runtime.h>
#include <hip/hip_bf16.h>

// ---------------------------------------------------------------------------
// BackboneDenoisingLayer for MI455X (gfx1250, wave32, WMMA).
// Dense GEMMs: v_wmma_f32_16x16x32_bf16, 64x128 block tile, 8 wave32 waves,
// each wave 32x32 (4 WMMA/chunk), double-buffered LDS, b128 global loads,
// b128 LDS fragment loads (B staged transposed).
// Segment-softmax attention uses f32 atomics (accumulators fit in 192MB L2).
// ---------------------------------------------------------------------------

#define N_NODES 8192
#define CS 256
#define CL 64
#define EIN 128
#define CZ 128
#define CH 16
#define H_HEADS 8
#define PQK 4
#define PV 8
#define E_EDGES 131072
#define ESEQ_EDGES 16384
#define PROJ_W 768        // q(128) | kv(256) | q_pts(96) | kv_pts(288)
#define FEAT_W 1408       // o(128) | o_pt(192) | o_pt_norm(64) | o_pair(1024)

typedef __attribute__((ext_vector_type(16))) __bf16 v16bf;
typedef __attribute__((ext_vector_type(8)))  __bf16 v8bf;
typedef __attribute__((ext_vector_type(8)))  float  v8f;

// f32 -> bf16 bits, round-to-nearest-even (integer path, toolchain-safe).
__device__ __forceinline__ unsigned bf16bits(float f) {
  union { float f; unsigned u; } v; v.f = f;
  return (v.u + 0x7fffu + ((v.u >> 16) & 1u)) >> 16;
}
__device__ __forceinline__ unsigned pack_bf16x2(float lo, float hi) {
  return bf16bits(lo) | (bf16bits(hi) << 16);
}

__device__ __forceinline__ void atomicMaxFloat(float* addr, float val) {
  unsigned* ia = (unsigned*)addr;
  unsigned old = *ia;
  while (true) {
    float f = __uint_as_float(old);
    if (f >= val) break;
    unsigned assumed = old;
    old = atomicCAS(ia, assumed, __float_as_uint(val));
    if (old == assumed) break;
  }
}

// ---------------------------------------------------------------------------
// WMMA GEMM: C[M x Ncols] = act(A[M x K] * B[K x Ncols] + bias + add)
// Requirements: M % 64 == 0, K % 32 == 0 (true for all call sites).
// Ncols arbitrary (guarded). Block tile 64x128; wave (2x4 grid) owns 32x32.
// ---------------------------------------------------------------------------
#define TM 64
#define TN 128
#define KC 32
#define LDK (KC + 8)      // 40 elems * 2B = 80B row stride (16B multiple)

__global__ __launch_bounds__(256)
void gemm_wmma_kernel(const float* __restrict__ A, int lda,
                      const float* __restrict__ B, int ldb,
                      const float* __restrict__ bias,
                      const float* __restrict__ add, int ldadd,
                      float* __restrict__ C, int ldc,
                      int M, int Ncols, int K, int act)
{
  __shared__ __align__(16) __bf16 As[2][TM][LDK];   // A row-major   (row, k)
  __shared__ __align__(16) __bf16 Bt[2][TN][LDK];   // B transposed  (col, k)

  const int tileM = blockIdx.y * TM;
  const int tileN = blockIdx.x * TN;
  const int tid   = threadIdx.x;
  const int wave  = tid >> 5;
  const int lane  = tid & 31;
  const int wm    = wave >> 2;   // 0..1 -> rows [wm*32, +32)
  const int wn    = wave & 3;    // 0..3 -> cols [wn*32, +32)
  const int hl    = lane >> 4;   // lane half
  const int l16   = lane & 15;

  v8f acc[2][2] = {};

  // ---- staging helper (chunk k0 into buffer buf) ----
  auto stage = [&](int k0, int buf) {
    // A: 64x32 floats, 2048 elems -> 8/thread as two float4 (b128 loads).
    #pragma unroll
    for (int seg = 0; seg < 2; ++seg) {
      int idx = tid + seg * 256;          // 0..511
      int r = idx >> 3;                   // 0..63
      int c = (idx & 7) * 4;              // 0,4,...,28
      const float4 v = *(const float4*)&A[(size_t)(tileM + r) * lda + k0 + c];
      *(uint2*)&As[buf][r][c] =
          make_uint2(pack_bf16x2(v.x, v.y), pack_bf16x2(v.z, v.w));
    }
    // B: 32x128 floats staged transposed. Thread owns column n, 4 k's.
    #pragma unroll
    for (int seg = 0; seg < 4; ++seg) {
      int idx = tid + seg * 256;          // 0..1023
      int n = idx & 127;
      int kq = idx >> 7;                  // 0..7 -> k = kq*4..+3
      int gc = tileN + n;
      float v0 = 0.f, v1 = 0.f, v2 = 0.f, v3 = 0.f;
      if (gc < Ncols) {
        const float* bp = B + (size_t)(k0 + kq * 4) * ldb + gc;
        v0 = bp[0];
        v1 = bp[(size_t)ldb];
        v2 = bp[(size_t)ldb * 2];
        v3 = bp[(size_t)ldb * 3];
      }
      *(uint2*)&Bt[buf][n][kq * 4] =
          make_uint2(pack_bf16x2(v0, v1), pack_bf16x2(v2, v3));
    }
  };

  stage(0, 0);
  const int nchunks = K / KC;
  int buf = 0;

  for (int kc = 0; kc < nchunks; ++kc) {
    __syncthreads();
    if (kc + 1 < nchunks) stage((kc + 1) * KC, buf ^ 1);

    // Fragments. A 16x32 ISA layout: lanes 0-15 elems 0..7 -> K0..7,
    // elems 8..15 -> K16..23; lanes 16-31 shifted by 8/8.
    v16bf afr[2], bfr[2];
    #pragma unroll
    for (int i = 0; i < 2; ++i) {
      const __bf16* ap = &As[buf][wm * 32 + i * 16 + l16][0];
      v8bf lo = *(const v8bf*)(ap + hl * 8);
      v8bf hi = *(const v8bf*)(ap + 16 + hl * 8);
      afr[i] = __builtin_shufflevector(lo, hi, 0, 1, 2, 3, 4, 5, 6, 7,
                                       8, 9, 10, 11, 12, 13, 14, 15);
    }
    // B 32x16: same K order as A (lanes 0-15 K0..7,K16..23; lanes 16-31 K8..15,K24..31).
    #pragma unroll
    for (int j = 0; j < 2; ++j) {
      const __bf16* bp = &Bt[buf][wn * 32 + j * 16 + l16][0];
      v8bf lo = *(const v8bf*)(bp + hl * 8);
      v8bf hi = *(const v8bf*)(bp + 16 + hl * 8);
      bfr[j] = __builtin_shufflevector(lo, hi, 0, 1, 2, 3, 4, 5, 6, 7,
                                       8, 9, 10, 11, 12, 13, 14, 15);
    }
    #pragma unroll
    for (int i = 0; i < 2; ++i)
      #pragma unroll
      for (int j = 0; j < 2; ++j)
        acc[i][j] = __builtin_amdgcn_wmma_f32_16x16x32_bf16(
            false, afr[i], false, bfr[j], (short)0, acc[i][j], false, false);

    buf ^= 1;
  }

  // Epilogue. C/D layout: lane half hl, VGPR r -> row = hl*8+r, col = l16.
  #pragma unroll
  for (int i = 0; i < 2; ++i) {
    #pragma unroll
    for (int r = 0; r < 8; ++r) {
      int row = tileM + wm * 32 + i * 16 + hl * 8 + r;
      #pragma unroll
      for (int j = 0; j < 2; ++j) {
        int col = tileN + wn * 32 + j * 16 + l16;
        if (col >= Ncols) continue;
        float v = acc[i][j][r];
        if (bias) v += bias[col];
        if (add)  v += add[(size_t)row * ldadd + col];
        if (act == 1) v = fmaxf(v, 0.0f);
        C[(size_t)row * ldc + col] = v;
      }
    }
  }
}

// ---------------------------------------------------------------------------
// Row LayerNorm: out = LN(x + add*maskRow) * g + b.  One block per row.
// ---------------------------------------------------------------------------
__global__ void ln_kernel(float* __restrict__ out, const float* __restrict__ x,
                          const float* __restrict__ add,
                          const float* __restrict__ maskRow,
                          const float* __restrict__ g, const float* __restrict__ b,
                          int D)
{
  __shared__ float sh[256];
  __shared__ float s_mu, s_rstd;
  const int row = blockIdx.x;
  const int t = threadIdx.x;
  float mr = maskRow ? maskRow[row] : 1.0f;
  float val = x[(size_t)row * D + t];
  if (add) val += add[(size_t)row * D + t] * mr;
  sh[t] = val; __syncthreads();
  for (int off = D >> 1; off > 0; off >>= 1) {
    if (t < off) sh[t] += sh[t + off];
    __syncthreads();
  }
  if (t == 0) s_mu = sh[0] / (float)D;
  __syncthreads();
  float mu = s_mu;
  float d = val - mu;
  sh[t] = d * d; __syncthreads();
  for (int off = D >> 1; off > 0; off >>= 1) {
    if (t < off) sh[t] += sh[t + off];
    __syncthreads();
  }
  if (t == 0) s_rstd = rsqrtf(sh[0] / (float)D + 1e-5f);
  __syncthreads();
  out[(size_t)row * D + t] = d * s_rstd * g[t] + b[t];
}

// ---------------------------------------------------------------------------
// Small helpers
// ---------------------------------------------------------------------------
__global__ void fill_kernel(float* p, float v, long long n) {
  long long i = (long long)blockIdx.x * 256 + threadIdx.x;
  if (i < n) p[i] = v;
}

__global__ void concat_kernel(const float* __restrict__ node,
                              const float* __restrict__ latent,
                              float* __restrict__ out, int Nn) {
  long long i = (long long)blockIdx.x * 256 + threadIdx.x;
  if (i >= (long long)Nn * 320) return;
  int n = (int)(i / 320), c = (int)(i % 320);
  out[i] = (c < CS) ? node[(size_t)n * CS + c] : latent[(size_t)n * CL + (c - CS)];
}

__global__ void rowscale_kernel(float* __restrict__ out, const float* __restrict__ in,
                                const float* __restrict__ scale, int D, long long total) {
  long long i = (long long)blockIdx.x * 256 + threadIdx.x;
  if (i >= total) return;
  int row = (int)(i / D);
  out[i] = in[i] * scale[row];
}

// Rotate + translate the 128 projected points per node (in place in proj).
__global__ void frame_points_kernel(float* __restrict__ proj,
                                    const float* __restrict__ rot,
                                    const float* __restrict__ trans, int Nn) {
  long long i = (long long)blockIdx.x * 256 + threadIdx.x;
  if (i >= (long long)Nn * 128) return;
  int n = (int)(i >> 7), pt = (int)(i & 127);
  int cb = (pt < 32) ? (384 + pt * 3) : (480 + (pt - 32) * 3);
  float* p = proj + (size_t)n * PROJ_W + cb;
  const float* R = rot + (size_t)n * 9;
  const float* T = trans + (size_t)n * 3;
  float v0 = p[0], v1 = p[1], v2 = p[2];
  float o0 = R[0] * v0 + R[1] * v1 + R[2] * v2 + T[0];
  float o1 = R[3] * v0 + R[4] * v1 + R[5] * v2 + T[1];
  float o2 = R[6] * v0 + R[7] * v1 + R[8] * v2 + T[2];
  p[0] = o0; p[1] = o1; p[2] = o2;
}

// Per-(edge,head) logits + segment max.
__global__ void logits_kernel(const float* __restrict__ proj,
                              const float* __restrict__ z,
                              const int* __restrict__ eidx, int Ecnt,
                              const float* __restrict__ wb, const float* __restrict__ bb,
                              const float* __restrict__ hw,
                              const float* __restrict__ mask,
                              float* __restrict__ logit, float* __restrict__ mmax)
{
  long long t = (long long)blockIdx.x * 256 + threadIdx.x;
  if (t >= (long long)Ecnt * H_HEADS) return;
  int e = (int)(t >> 3), h = (int)(t & 7);
  int dst = eidx[e];
  int src = eidx[Ecnt + e];
  const float* pd = proj + (size_t)dst * PROJ_W;
  const float* ps = proj + (size_t)src * PROJ_W;
  float qk = 0.f;
  #pragma unroll
  for (int c = 0; c < CH; ++c) qk += pd[h * 16 + c] * ps[128 + h * 32 + c];
  qk *= 0.14433756729740643f;              // 1/sqrt(3*CH)
  float ssq = 0.f;
  #pragma unroll
  for (int p = 0; p < PQK; ++p)
    #pragma unroll
    for (int i = 0; i < 3; ++i) {
      float d = pd[384 + h * 12 + p * 3 + i] - ps[480 + h * 36 + p * 3 + i];
      ssq += d * d;
    }
  float headw = log1pf(expf(hw[h]));       // softplus
  float pt_att = -0.5f * 0.13608276348795434f * ssq * headw;  // gamma=sqrt(1/54)
  float bpair = bb[h];
  const float* zr = z + (size_t)e * CZ;
  for (int c = 0; c < CZ; ++c) bpair += zr[c] * wb[c * H_HEADS + h];
  float mb = 100000.0f * (mask[src] * mask[dst] - 1.0f);
  float l = qk + 0.5773502691896258f * bpair + pt_att + mb;
  logit[(size_t)e * H_HEADS + h] = l;
  atomicMaxFloat(&mmax[(size_t)dst * H_HEADS + h], l);
}

__global__ void expden_kernel(float* __restrict__ logit,
                              const int* __restrict__ eidx, int Ecnt,
                              const float* __restrict__ mmax,
                              float* __restrict__ den)
{
  long long t = (long long)blockIdx.x * 256 + threadIdx.x;
  if (t >= (long long)Ecnt * H_HEADS) return;
  int e = (int)(t >> 3), h = (int)(t & 7);
  int dst = eidx[e];
  float m = mmax[(size_t)dst * H_HEADS + h];
  if (m < -1e29f) m = 0.f;                 // jnp.where(isfinite(m), m, 0)
  float ev = expf(logit[t] - m);
  logit[t] = ev;
  atomicAdd(&den[(size_t)dst * H_HEADS + h], ev);
}

// One block per edge: scatter o / o_pt / o_pair with f32 atomics (L2-resident).
__global__ __launch_bounds__(256)
void accum_kernel(const float* __restrict__ proj, const float* __restrict__ z,
                  const int* __restrict__ eidx, int Ecnt,
                  const float* __restrict__ logit, const float* __restrict__ den,
                  float* __restrict__ feats, float* __restrict__ optbuf)
{
  const int e = blockIdx.x;
  const int dst = eidx[e];
  const int src = eidx[Ecnt + e];
  __shared__ float at[H_HEADS];
  if (threadIdx.x < H_HEADS) {
    int h = threadIdx.x;
    at[h] = logit[(size_t)e * H_HEADS + h] /
            (den[(size_t)dst * H_HEADS + h] + 1e-9f);
  }
  __syncthreads();
  const float* ps = proj + (size_t)src * PROJ_W;
  const float* zr = z + (size_t)e * CZ;
  float* fd = feats + (size_t)dst * FEAT_W;
  for (int i = threadIdx.x; i < 1344; i += 256) {
    if (i < 128) {                          // o: (h, c)
      int h = i >> 4, c = i & 15;
      atomicAdd(&fd[i], at[h] * ps[128 + h * 32 + 16 + c]);
    } else if (i < 320) {                   // o_pt raw: (h, p, xyz)
      int j = i - 128;
      int h = j / 24, rem = j % 24;
      atomicAdd(&optbuf[(size_t)dst * 192 + j],
                at[h] * ps[480 + h * 36 + 12 + rem]);
    } else {                                // o_pair: (h, cz)
      int j = i - 320;
      int h = j >> 7, c = j & 127;
      atomicAdd(&fd[384 + j], at[h] * zr[c]);
    }
  }
}

// Inverse-frame transform of o_pt + norms, scattered into feats layout
// [o | o_pt(i-major,h,p) | o_pt_norm(h,p) | o_pair].
__global__ void finalize_pts_kernel(const float* __restrict__ optbuf,
                                    const float* __restrict__ rot,
                                    const float* __restrict__ trans,
                                    float* __restrict__ feats, int Nn)
{
  long long t = (long long)blockIdx.x * 256 + threadIdx.x;
  if (t >= (long long)Nn * 64) return;
  int n = (int)(t >> 6), hp = (int)(t & 63);
  int h = hp >> 3, p = hp & 7;
  const float* R = rot + (size_t)n * 9;
  const float* T = trans + (size_t)n * 3;
  const float* o = optbuf + (size_t)n * 192 + h * 24 + p * 3;
  float d0 = o[0] - T[0], d1 = o[1] - T[1], d2 = o[2] - T[2];
  // einsum('nji,...j->...i'): out_i = sum_j R[j][i] * d_j  (R transpose)
  float o0 = R[0] * d0 + R[3] * d1 + R[6] * d2;
  float o1 = R[1] * d0 + R[4] * d1 + R[7] * d2;
  float o2 = R[2] * d0 + R[5] * d1 + R[8] * d2;
  float* f = feats + (size_t)n * FEAT_W;
  f[128 + 0 * 64 + h * 8 + p] = o0;
  f[128 + 1 * 64 + h * 8 + p] = o1;
  f[128 + 2 * 64 + h * 8 + p] = o2;
  f[320 + h * 8 + p] = sqrtf(o0 * o0 + o1 * o1 + o2 * o2 + 1e-8f);
}

// upd = (s*noise) @ bb_w + bb_b   (N x 256 x 6, too narrow for WMMA)
__global__ void bb_kernel(const float* __restrict__ sn, const float* __restrict__ w,
                          const float* __restrict__ b, float* __restrict__ upd, int Nn)
{
  int n = blockIdx.x * 256 + threadIdx.x;
  if (n >= Nn) return;
  float acc[6];
  #pragma unroll
  for (int j = 0; j < 6; ++j) acc[j] = b[j];
  const float* sr = sn + (size_t)n * CS;
  for (int k = 0; k < CS; ++k) {
    float sv = sr[k];
    #pragma unroll
    for (int j = 0; j < 6; ++j) acc[j] += sv * w[k * 6 + j];
  }
  #pragma unroll
  for (int j = 0; j < 6; ++j) upd[(size_t)n * 6 + j] = acc[j];
}

__global__ void compose_kernel(const float* __restrict__ rot,
                               const float* __restrict__ trans,
                               const float* __restrict__ upd,
                               const float* __restrict__ noise,
                               float* __restrict__ rotO, float* __restrict__ transO,
                               int Nn)
{
  int n = blockIdx.x * 256 + threadIdx.x;
  if (n >= Nn) return;
  float nz = noise[n];
  float qx = upd[(size_t)n * 6 + 0] * nz;
  float qy = upd[(size_t)n * 6 + 1] * nz;
  float qz = upd[(size_t)n * 6 + 2] * nz;
  float t0 = upd[(size_t)n * 6 + 3] * nz;
  float t1 = upd[(size_t)n * 6 + 4] * nz;
  float t2 = upd[(size_t)n * 6 + 5] * nz;
  float inv = rsqrtf(1.0f + qx * qx + qy * qy + qz * qz);
  float w = inv, x = qx * inv, y = qy * inv, z = qz * inv;
  float ru[9] = {
    1 - 2 * (y * y + z * z), 2 * (x * y - w * z),     2 * (x * z + w * y),
    2 * (x * y + w * z),     1 - 2 * (x * x + z * z), 2 * (y * z - w * x),
    2 * (x * z - w * y),     2 * (y * z + w * x),     1 - 2 * (x * x + y * y)};
  const float* R = rot + (size_t)n * 9;
  #pragma unroll
  for (int i = 0; i < 3; ++i)
    #pragma unroll
    for (int k = 0; k < 3; ++k)
      rotO[(size_t)n * 9 + i * 3 + k] =
          R[i * 3 + 0] * ru[0 * 3 + k] + R[i * 3 + 1] * ru[1 * 3 + k] +
          R[i * 3 + 2] * ru[2 * 3 + k];
  #pragma unroll
  for (int i = 0; i < 3; ++i)
    transO[(size_t)n * 3 + i] = trans[(size_t)n * 3 + i] +
        R[i * 3 + 0] * t0 + R[i * 3 + 1] * t1 + R[i * 3 + 2] * t2;
}

// ---------------------------------------------------------------------------
// Host orchestration
// ---------------------------------------------------------------------------
static inline void gemm(hipStream_t st, const float* A, int lda,
                        const float* B, int ldb, const float* bias,
                        const float* add, int ldadd, float* C, int ldc,
                        int M, int N, int K, int act) {
  dim3 grid((N + TN - 1) / TN, M / TM);   // M % 64 == 0 at every call site
  gemm_wmma_kernel<<<grid, 256, 0, st>>>(A, lda, B, ldb, bias, add, ldadd,
                                         C, ldc, M, N, K, act);
}

static inline void fillf(hipStream_t st, float* p, float v, long long n) {
  fill_kernel<<<(unsigned)((n + 255) / 256), 256, 0, st>>>(p, v, n);
}

struct IpaP {
  const float *wq, *bq, *wkv, *bkv, *wqp, *bqp, *wkvp, *bkvp;
  const float *wb, *bb, *hw, *wout, *bout;
};

static void run_ipa(hipStream_t st, const float* s, const float* z,
                    const int* eidx, int Ecnt, const float* rot,
                    const float* trans, const float* mask, const IpaP& P,
                    float* proj, float* logit, float* mmax, float* den,
                    float* feats, float* optbuf, float* u, int Nn)
{
  // Node projections (column-offset writes into one N x 768 buffer).
  gemm(st, s, CS, P.wq,   128, P.bq,   nullptr, 0, proj + 0,   PROJ_W, Nn, 128, CS, 0);
  gemm(st, s, CS, P.wkv,  256, P.bkv,  nullptr, 0, proj + 128, PROJ_W, Nn, 256, CS, 0);
  gemm(st, s, CS, P.wqp,   96, P.bqp,  nullptr, 0, proj + 384, PROJ_W, Nn,  96, CS, 0);
  gemm(st, s, CS, P.wkvp, 288, P.bkvp, nullptr, 0, proj + 480, PROJ_W, Nn, 288, CS, 0);
  frame_points_kernel<<<(Nn * 128 + 255) / 256, 256, 0, st>>>(proj, rot, trans, Nn);

  fillf(st, mmax, -1e30f, (long long)Nn * H_HEADS);
  fillf(st, den, 0.0f, (long long)Nn * H_HEADS);
  fillf(st, feats, 0.0f, (long long)Nn * FEAT_W);
  fillf(st, optbuf, 0.0f, (long long)Nn * 192);

  int eh_blocks = (Ecnt * H_HEADS + 255) / 256;
  logits_kernel<<<eh_blocks, 256, 0, st>>>(proj, z, eidx, Ecnt, P.wb, P.bb,
                                           P.hw, mask, logit, mmax);
  expden_kernel<<<eh_blocks, 256, 0, st>>>(logit, eidx, Ecnt, mmax, den);
  accum_kernel<<<Ecnt, 256, 0, st>>>(proj, z, eidx, Ecnt, logit, den, feats, optbuf);
  finalize_pts_kernel<<<(Nn * 64 + 255) / 256, 256, 0, st>>>(optbuf, rot, trans,
                                                             feats, Nn);
  // u = feats @ wout + bout
  gemm(st, feats, FEAT_W, P.wout, CS, P.bout, nullptr, 0, u, CS, Nn, CS, FEAT_W, 0);
}

static IpaP load_ipa(void* const* d_in, int base) {
  IpaP p;
  p.wq   = (const float*)d_in[base + 0];  p.bq   = (const float*)d_in[base + 1];
  p.wkv  = (const float*)d_in[base + 2];  p.bkv  = (const float*)d_in[base + 3];
  p.wqp  = (const float*)d_in[base + 4];  p.bqp  = (const float*)d_in[base + 5];
  p.wkvp = (const float*)d_in[base + 6];  p.bkvp = (const float*)d_in[base + 7];
  p.wb   = (const float*)d_in[base + 8];  p.bb   = (const float*)d_in[base + 9];
  p.hw   = (const float*)d_in[base + 10];
  p.wout = (const float*)d_in[base + 11]; p.bout = (const float*)d_in[base + 12];
  return p;
}

extern "C" void kernel_launch(void* const* d_in, const int* in_sizes, int n_in,
                              void* d_out, int out_size, void* d_ws, size_t ws_size,
                              hipStream_t stream) {
  (void)in_sizes; (void)n_in; (void)out_size; (void)ws_size;
  // ---- inputs (setup_inputs dict order; params flattened depth-first) ----
  const float* node   = (const float*)d_in[0];
  const float* latent = (const float*)d_in[1];
  const float* edgef  = (const float*)d_in[2];
  const float* seqz   = (const float*)d_in[3];
  const float* rot    = (const float*)d_in[4];
  const float* trans  = (const float*)d_in[5];
  const float* mask   = (const float*)d_in[6];
  const float* noise  = (const float*)d_in[7];
  const int*   eidx   = (const int*)d_in[8];
  const int*   seidx  = (const int*)d_in[9];
  // params: 10..17 edge_embed{w1,b1,w2,b2,w3,b3,g,be}, 18..19 nu_{w,b},
  // 20..32 ipa1, 33..45 ipa2, 46..47 ln1_{g,b}, 48..55 nt{w1..b3,g,b},
  // 56..57 bb_{w,b}, 58..59 lat_{w,b}
  const float* ee_w1 = (const float*)d_in[10]; const float* ee_b1 = (const float*)d_in[11];
  const float* ee_w2 = (const float*)d_in[12]; const float* ee_b2 = (const float*)d_in[13];
  const float* ee_w3 = (const float*)d_in[14]; const float* ee_b3 = (const float*)d_in[15];
  const float* ee_g  = (const float*)d_in[16]; const float* ee_be = (const float*)d_in[17];
  const float* nu_w  = (const float*)d_in[18]; const float* nu_b  = (const float*)d_in[19];
  IpaP ipa1 = load_ipa(d_in, 20);
  IpaP ipa2 = load_ipa(d_in, 33);
  const float* ln1_g = (const float*)d_in[46]; const float* ln1_b = (const float*)d_in[47];
  const float* nt_w1 = (const float*)d_in[48]; const float* nt_b1 = (const float*)d_in[49];
  const float* nt_w2 = (const float*)d_in[50]; const float* nt_b2 = (const float*)d_in[51];
  const float* nt_w3 = (const float*)d_in[52]; const float* nt_b3 = (const float*)d_in[53];
  const float* nt_g  = (const float*)d_in[54]; const float* nt_b  = (const float*)d_in[55];
  const float* bb_w  = (const float*)d_in[56]; const float* bb_b  = (const float*)d_in[57];
  const float* lat_w = (const float*)d_in[58]; const float* lat_b = (const float*)d_in[59];

  // ---- workspace carve ----
  char* wp = (char*)d_ws;
  auto alloc = [&](size_t nfloats) {
    float* p = (float*)wp;
    wp += (nfloats * sizeof(float) + 255) & ~(size_t)255;
    return p;
  };
  float* zb0    = alloc((size_t)E_EDGES * CZ);
  float* zb1    = alloc((size_t)E_EDGES * CZ);
  float* s_in   = alloc((size_t)N_NODES * 320);
  float* s      = alloc((size_t)N_NODES * CS);
  float* u      = alloc((size_t)N_NODES * CS);
  float* proj   = alloc((size_t)N_NODES * PROJ_W);
  float* logit  = alloc((size_t)E_EDGES * H_HEADS);
  float* mmax   = alloc((size_t)N_NODES * H_HEADS);
  float* den    = alloc((size_t)N_NODES * H_HEADS);
  float* feats  = alloc((size_t)N_NODES * FEAT_W);
  float* optbuf = alloc((size_t)N_NODES * 192);
  float* h1     = alloc((size_t)N_NODES * CS);
  float* h2     = alloc((size_t)N_NODES * CS);
  float* upd    = alloc((size_t)N_NODES * 6);

  // ---- output regions (tuple: s, rot, trans, latent) ----
  float* out_s      = (float*)d_out;
  float* out_rot    = out_s + (size_t)N_NODES * CS;
  float* out_trans  = out_rot + (size_t)N_NODES * 9;
  float* out_latent = out_trans + (size_t)N_NODES * 3;

  // 1) Edge embed MLP + LN -> z (zb0)
  gemm(stream, edgef, EIN, ee_w1, CZ, ee_b1, nullptr, 0, zb1, CZ, E_EDGES, CZ, EIN, 1);
  gemm(stream, zb1,   CZ,  ee_w2, CZ, ee_b2, nullptr, 0, zb0, CZ, E_EDGES, CZ, CZ,  1);
  gemm(stream, zb0,   CZ,  ee_w3, CZ, ee_b3, nullptr, 0, zb1, CZ, E_EDGES, CZ, CZ,  0);
  ln_kernel<<<E_EDGES, CZ, 0, stream>>>(zb0, zb1, nullptr, nullptr, ee_g, ee_be, CZ);

  // 2) s = concat(node, latent) @ nu_w + nu_b
  concat_kernel<<<((long long)N_NODES * 320 + 255) / 256, 256, 0, stream>>>(
      node, latent, s_in, N_NODES);
  gemm(stream, s_in, 320, nu_w, CS, nu_b, nullptr, 0, s, CS, N_NODES, CS, 320, 0);

  // 3) IPA-1 (graph edges) -> u;  s = LN(s + u*mask)
  run_ipa(stream, s, zb0, eidx, E_EDGES, rot, trans, mask, ipa1,
          proj, logit, mmax, den, feats, optbuf, u, N_NODES);
  ln_kernel<<<N_NODES, CS, 0, stream>>>(s, s, u, mask, ln1_g, ln1_b, CS);

  // 4) IPA-2 (sequence edges, z = seq_edge_features) -> u;  s = LN(s + u*mask)
  run_ipa(stream, s, seqz, seidx, ESEQ_EDGES, rot, trans, mask, ipa2,
          proj, logit, mmax, den, feats, optbuf, u, N_NODES);
  ln_kernel<<<N_NODES, CS, 0, stream>>>(s, s, u, mask, ln1_g, ln1_b, CS);

  // 5) Node transition MLP + LN
  gemm(stream, s,  CS, nt_w1, CS, nt_b1, nullptr, 0, h1, CS, N_NODES, CS, CS, 1);
  gemm(stream, h1, CS, nt_w2, CS, nt_b2, nullptr, 0, h2, CS, N_NODES, CS, CS, 1);
  gemm(stream, h2, CS, nt_w3, CS, nt_b3, nullptr, 0, h1, CS, N_NODES, CS, CS, 0);
  ln_kernel<<<N_NODES, CS, 0, stream>>>(s, s, h1, nullptr, nt_g, nt_b, CS);

  // 6) s *= res_mask  -> output region (also feeds upd/latent)
  rowscale_kernel<<<((long long)N_NODES * CS + 255) / 256, 256, 0, stream>>>(
      out_s, s, mask, CS, (long long)N_NODES * CS);

  // 7) Backbone update: upd = (s*noise) @ bb_w + bb_b; compose frames
  rowscale_kernel<<<((long long)N_NODES * CS + 255) / 256, 256, 0, stream>>>(
      h2, out_s, noise, CS, (long long)N_NODES * CS);
  bb_kernel<<<(N_NODES + 255) / 256, 256, 0, stream>>>(h2, bb_w, bb_b, upd, N_NODES);
  compose_kernel<<<(N_NODES + 255) / 256, 256, 0, stream>>>(
      rot, trans, upd, noise, out_rot, out_trans, N_NODES);

  // 8) latent_out = latent + s @ lat_w + lat_b  (WMMA with fused residual add)
  gemm(stream, out_s, CS, lat_w, CL, lat_b, latent, CL, out_latent, CL,
       N_NODES, CL, CS, 0);
}
